// CausalDiTBlock_12438225290012
// MI455X (gfx1250) — matrix operand order repaired and checked by reading, not compiled, not run on hardware
//
#include <hip/hip_runtime.h>

// ---------------------------------------------------------------------------
// Causal DiT block forward for MI455X (gfx1250, wave32, WMMA bf16).
// All matmuls run through v_wmma_f32_16x16x32_bf16 with fp32 accumulation.
// Tile staging uses CDNA5 async global->LDS copies (ASYNCcnt) with double
// buffering in the GEMM. Shapes hardcoded to the reference:
// B=1, T=3072 (6x512), D=1536, H=12, hd=128, FF=8960, Tc=512.
// ---------------------------------------------------------------------------

typedef __attribute__((ext_vector_type(16))) __bf16 bf16x16;
typedef __attribute__((ext_vector_type(8)))  float  f32x8;

#define D_MODEL   1536
#define T_SEQ     3072
#define T_CTX     512
#define FFN_DIM   8960
#define N_HEADS   12
#define HEAD_DIM  128
#define FRAME     512

// ---------------------------------------------------------------------------
// CDNA5 async global->LDS copy (tracked by ASYNCcnt, bypasses VGPRs).
// LDS destination is the wave-relative byte offset = low 32 bits of the
// generic pointer (flat aperture truncation rule).
// ---------------------------------------------------------------------------
__device__ inline void async_load_b128(const void* lds_ptr, const void* gaddr) {
  unsigned lds_off = (unsigned)(unsigned long long)(uintptr_t)lds_ptr;
  unsigned long long ga = (unsigned long long)(uintptr_t)gaddr;
  asm volatile("global_load_async_to_lds_b128 %0, %1, off"
               :: "v"(lds_off), "v"(ga)
               : "memory");
}
__device__ inline void wait_async0() {
  asm volatile("s_wait_asynccnt 0x0" ::: "memory");
}

// ---------------------------------------------------------------------------
// WMMA helpers (CDNA5 16x16x32 bf16, fp32 accum)
// ---------------------------------------------------------------------------
__device__ inline f32x8 wmma_bf16(bf16x16 a, bf16x16 b, f32x8 c) {
  return __builtin_amdgcn_wmma_f32_16x16x32_bf16(
      /*neg_a=*/false, a, /*neg_b=*/false, b,
      /*c_mod=*/(short)0, c, /*reuse_a=*/false, /*reuse_b=*/false);
}

// A fragment (16x32, M x K). Storage: row-major, K contiguous, `stride` elems.
// Lane L holds row (row_base + L%16); kk = 16*(L/16) + {0,2,4,6} and {8,10,12,14}.
__device__ inline bf16x16 load_frag_a(const __bf16* sm, int stride,
                                      int row_base, int kbase, int lane) {
  union { bf16x16 v; unsigned u[8]; } r;
  int hg = lane >> 4;
  const __bf16* p = sm + (size_t)(row_base + (lane & 15)) * stride + kbase + 16 * hg;
  const unsigned* q0 = (const unsigned*)(p);
  const unsigned* q1 = (const unsigned*)(p + 8);
#pragma unroll
  for (int v = 0; v < 4; ++v) { r.u[v] = q0[v]; r.u[v + 4] = q1[v]; }
  return r.v;
}

// B fragment (32x16, K x N), with B stored transposed: N rows, K contiguous.
// Lane L holds column (col_base + L%16); kk = 16*(L/16) + 2v.
__device__ inline bf16x16 load_frag_b(const __bf16* sm, int stride,
                                      int col_base, int kbase, int lane) {
  union { bf16x16 v; unsigned u[8]; } r;
  int hg = lane >> 4;
  const unsigned* q =
      (const unsigned*)(sm + (size_t)(col_base + (lane & 15)) * stride + kbase + 16 * hg);
#pragma unroll
  for (int v = 0; v < 8; ++v) r.u[v] = q[v];
  return r.v;
}

__device__ inline float wave_reduce_sum(float v) {
#pragma unroll
  for (int m = 16; m >= 1; m >>= 1) v += __shfl_xor(v, m, 32);
  return v;
}

// ---------------------------------------------------------------------------
// Weight pack: fp32 (K x N) -> bf16 transposed (N x K)
// ---------------------------------------------------------------------------
__global__ __launch_bounds__(256) void transpose_to_bf16(
    const float* __restrict__ W, __bf16* __restrict__ Bt, int Kdim, int Ndim) {
  size_t idx = (size_t)blockIdx.x * 256 + threadIdx.x;
  size_t total = (size_t)Kdim * Ndim;
  if (idx >= total) return;
  int n = (int)(idx / Kdim);
  int k = (int)(idx % Kdim);
  Bt[idx] = (__bf16)W[(size_t)k * Ndim + n];
}

__global__ __launch_bounds__(256) void to_bf16_kernel(
    const float* __restrict__ src, __bf16* __restrict__ dst, size_t n) {
  size_t idx = (size_t)blockIdx.x * 256 + threadIdx.x;
  if (idx < n) dst[idx] = (__bf16)src[idx];
}

// ---------------------------------------------------------------------------
// LayerNorm (no affine) + per-frame AdaLN modulation -> bf16
// ---------------------------------------------------------------------------
__global__ __launch_bounds__(256) void ln_mod_kernel(
    const float* __restrict__ x, const float* __restrict__ modulation,
    const float* __restrict__ t_mod, __bf16* __restrict__ out,
    int i_shift, int i_scale) {
  int t = blockIdx.x;
  int f = t >> 9;  // /512
  const float* xt = x + (size_t)t * D_MODEL;
  float vals[6];
  float s1 = 0.f, s2 = 0.f;
#pragma unroll
  for (int i = 0; i < 6; ++i) {
    float v = xt[threadIdx.x + i * 256];
    vals[i] = v; s1 += v; s2 += v * v;
  }
  __shared__ float sh1[8], sh2[8];
  int lane = threadIdx.x & 31, wv = threadIdx.x >> 5;
  float w1 = wave_reduce_sum(s1), w2 = wave_reduce_sum(s2);
  if (lane == 0) { sh1[wv] = w1; sh2[wv] = w2; }
  __syncthreads();
  float t1 = 0.f, t2 = 0.f;
#pragma unroll
  for (int i = 0; i < 8; ++i) { t1 += sh1[i]; t2 += sh2[i]; }
  float mean = t1 * (1.f / D_MODEL);
  float var  = t2 * (1.f / D_MODEL) - mean * mean;
  float rstd = rsqrtf(var + 1e-6f);
  const float* msh = modulation + (size_t)i_shift * D_MODEL;
  const float* msc = modulation + (size_t)i_scale * D_MODEL;
  const float* tsh = t_mod + ((size_t)f * 6 + i_shift) * D_MODEL;
  const float* tsc = t_mod + ((size_t)f * 6 + i_scale) * D_MODEL;
  __bf16* ot = out + (size_t)t * D_MODEL;
#pragma unroll
  for (int i = 0; i < 6; ++i) {
    int d = threadIdx.x + i * 256;
    float xhat = (vals[i] - mean) * rstd;
    ot[d] = (__bf16)(xhat * (1.f + msc[d] + tsc[d]) + msh[d] + tsh[d]);
  }
}

// LayerNorm with affine (norm3) -> bf16
__global__ __launch_bounds__(256) void ln_affine_kernel(
    const float* __restrict__ x, const float* __restrict__ w,
    const float* __restrict__ b, __bf16* __restrict__ out) {
  int t = blockIdx.x;
  const float* xt = x + (size_t)t * D_MODEL;
  float vals[6];
  float s1 = 0.f, s2 = 0.f;
#pragma unroll
  for (int i = 0; i < 6; ++i) {
    float v = xt[threadIdx.x + i * 256];
    vals[i] = v; s1 += v; s2 += v * v;
  }
  __shared__ float sh1[8], sh2[8];
  int lane = threadIdx.x & 31, wv = threadIdx.x >> 5;
  float w1 = wave_reduce_sum(s1), w2 = wave_reduce_sum(s2);
  if (lane == 0) { sh1[wv] = w1; sh2[wv] = w2; }
  __syncthreads();
  float t1 = 0.f, t2 = 0.f;
#pragma unroll
  for (int i = 0; i < 8; ++i) { t1 += sh1[i]; t2 += sh2[i]; }
  float mean = t1 * (1.f / D_MODEL);
  float var  = t2 * (1.f / D_MODEL) - mean * mean;
  float rstd = rsqrtf(var + 1e-6f);
  __bf16* ot = out + (size_t)t * D_MODEL;
#pragma unroll
  for (int i = 0; i < 6; ++i) {
    int d = threadIdx.x + i * 256;
    ot[d] = (__bf16)(((vals[i] - mean) * rstd) * w[d] + b[d]);
  }
}

// ---------------------------------------------------------------------------
// RMSNorm over D (optionally followed by interleaved RoPE per head) -> bf16
// ---------------------------------------------------------------------------
__global__ __launch_bounds__(256) void rms_rope_kernel(
    const float* __restrict__ y, const float* __restrict__ nw,
    const float* __restrict__ cosb, const float* __restrict__ sinb,
    __bf16* __restrict__ out, int do_rope) {
  int t = blockIdx.x;
  const float* yt = y + (size_t)t * D_MODEL;
  float s2 = 0.f;
#pragma unroll
  for (int i = 0; i < 6; ++i) {
    float v = yt[threadIdx.x + i * 256];
    s2 += v * v;
  }
  __shared__ float sh2[8];
  int lane = threadIdx.x & 31, wv = threadIdx.x >> 5;
  float w2 = wave_reduce_sum(s2);
  if (lane == 0) sh2[wv] = w2;
  __syncthreads();
  float t2 = 0.f;
#pragma unroll
  for (int i = 0; i < 8; ++i) t2 += sh2[i];
  float r = rsqrtf(t2 * (1.f / D_MODEL) + 1e-6f);
  __bf16* ot = out + (size_t)t * D_MODEL;
  if (do_rope) {
#pragma unroll
    for (int i = 0; i < 3; ++i) {
      int p = threadIdx.x + i * 256;  // 768 pairs total
      int head = p >> 6, pp = p & 63;
      int base = head * HEAD_DIM + 2 * pp;
      float xr = yt[base] * r * nw[base];
      float xi = yt[base + 1] * r * nw[base + 1];
      float c = cosb[(size_t)t * 64 + pp];
      float s = sinb[(size_t)t * 64 + pp];
      ot[base]     = (__bf16)(xr * c - xi * s);
      ot[base + 1] = (__bf16)(xr * s + xi * c);
    }
  } else {
#pragma unroll
    for (int i = 0; i < 6; ++i) {
      int d = threadIdx.x + i * 256;
      ot[d] = (__bf16)(yt[d] * r * nw[d]);
    }
  }
}

// ---------------------------------------------------------------------------
// Generic WMMA GEMM: C = A(bf16, MxK row-major) x Bt(bf16, NxK) + bias
// Block: 256 threads = 8 waves, tile 128x128, K step 32, double-buffered
// async global->LDS staging. Wave (wm,wn) owns 32x64 = 2x4 accumulators.
// EPI 0: fp32 out with bias. EPI 1: bias + tanh-GELU, bf16 out.
// Requires M%128==0, N%128==0, K%32==0 (true for all shapes used here).
// ---------------------------------------------------------------------------
template <int EPI>
__global__ __launch_bounds__(256) void gemm_bf16(
    const __bf16* __restrict__ A, const __bf16* __restrict__ Bt,
    const float* __restrict__ bias, void* __restrict__ Cout,
    int M, int N, int K) {
  __shared__ __bf16 As[2][128 * 32];
  __shared__ __bf16 Bs[2][128 * 32];
  int tid = threadIdx.x, lane = tid & 31, wv = tid >> 5;
  int wm = wv >> 1, wn = wv & 1;
  int bm = blockIdx.x * 128, bn = blockIdx.y * 128;
  (void)M;

  // staging map: thread -> (row, 16-elem column group); 2 x b128 per matrix
  int r = tid >> 1, c = (tid & 1) * 16;
  const __bf16* gA = A + (size_t)(bm + r) * K + c;
  const __bf16* gB = Bt + (size_t)(bn + r) * K + c;
  const __bf16* sA[2] = { As[0] + r * 32 + c, As[1] + r * 32 + c };
  const __bf16* sB[2] = { Bs[0] + r * 32 + c, Bs[1] + r * 32 + c };

  // prefetch first K tile into buffer 0
  async_load_b128(sA[0],     gA);
  async_load_b128(sA[0] + 8, gA + 8);
  async_load_b128(sB[0],     gB);
  async_load_b128(sB[0] + 8, gB + 8);
  wait_async0();
  __syncthreads();

  f32x8 acc[2][4] = {};
  int nk = K / 32;
  int p = 0;
  for (int ki = 0; ki < nk; ++ki) {
    // prefetch next K tile into the other buffer while computing this one
    if (ki + 1 < nk) {
      const __bf16* a2 = gA + (size_t)(ki + 1) * 32;
      const __bf16* b2 = gB + (size_t)(ki + 1) * 32;
      async_load_b128(sA[p ^ 1],     a2);
      async_load_b128(sA[p ^ 1] + 8, a2 + 8);
      async_load_b128(sB[p ^ 1],     b2);
      async_load_b128(sB[p ^ 1] + 8, b2 + 8);
    }
    bf16x16 af0 = load_frag_a(As[p], 32, wm * 32 + 0,  0, lane);
    bf16x16 af1 = load_frag_a(As[p], 32, wm * 32 + 16, 0, lane);
#pragma unroll
    for (int nt = 0; nt < 4; ++nt) {
      bf16x16 bf = load_frag_b(Bs[p], 32, wn * 64 + nt * 16, 0, lane);
      acc[0][nt] = wmma_bf16(af0, bf, acc[0][nt]);
      acc[1][nt] = wmma_bf16(af1, bf, acc[1][nt]);
    }
    wait_async0();
    __syncthreads();
    p ^= 1;
  }

  int hg = lane >> 4, l15 = lane & 15;
#pragma unroll
  for (int mt = 0; mt < 2; ++mt)
#pragma unroll
    for (int nt = 0; nt < 4; ++nt) {
      int col = bn + wn * 64 + nt * 16 + l15;
      float bvv = bias ? bias[col] : 0.f;
#pragma unroll
      for (int v = 0; v < 8; ++v) {
        int row = bm + wm * 32 + mt * 16 + v + 8 * hg;
        float val = acc[mt][nt][v] + bvv;
        if constexpr (EPI == 0) {
          ((float*)Cout)[(size_t)row * N + col] = val;
        } else {
          float u = 0.7978845608028654f * (val + 0.044715f * val * val * val);
          ((__bf16*)Cout)[(size_t)row * N + col] =
              (__bf16)(0.5f * val * (1.f + tanhf(u)));
        }
      }
    }
}

// ---------------------------------------------------------------------------
// Flash attention, one head per blockIdx.y, 64 query rows per blockIdx.x.
// 128 threads = 4 waves; wave w owns 16 query rows. WMMA for QK^T and PV.
// causal_block > 0 => frame-block-causal mask (kv limited to full frames).
// Q: [Tq x D], K/V: [Tk x D], head offset = head*128. Output bf16 [Tq x D].
// ---------------------------------------------------------------------------
__global__ __launch_bounds__(128) void attn_kernel(
    const __bf16* __restrict__ Q, const __bf16* __restrict__ K,
    const __bf16* __restrict__ V, __bf16* __restrict__ Y,
    int Tk, int causal_block) {
  __shared__ __bf16 Qs[64 * 128];
  __shared__ __bf16 Ks[64 * 128];
  __shared__ __bf16 Vts[128 * 64];   // transposed: [hd][key]
  __shared__ __bf16 Ps[4 * 16 * 64]; // per-wave P tile staging
  int tid = threadIdx.x, lane = tid & 31, wv = tid >> 5;
  int h = blockIdx.y;
  int q0 = blockIdx.x * 64;
  int hoff = h * HEAD_DIM;
  const float scale = 0.08838834764831845f;  // 1/sqrt(128)

  // stage Q tile (64 x 128) via async copies
#pragma unroll
  for (int i = 0; i < 8; ++i) {
    int u = tid + i * 128;
    int r = u >> 4, c = (u & 15) * 8;
    async_load_b128(Qs + r * 128 + c, Q + (size_t)(q0 + r) * D_MODEL + hoff + c);
  }
  wait_async0();
  __syncthreads();

  int kv_limit = causal_block ? ((q0 / causal_block) + 1) * causal_block : Tk;

  // hoist Q fragments (4 K-steps over hd=128)
  bf16x16 aq[4];
#pragma unroll
  for (int ks = 0; ks < 4; ++ks)
    aq[ks] = load_frag_a(Qs, 128, wv * 16, ks * 32, lane);

  f32x8 acc[8] = {};
  float m8[8], l8[8];
#pragma unroll
  for (int v = 0; v < 8; ++v) { m8[v] = -1e30f; l8[v] = 0.f; }

  int hg = lane >> 4, l15 = lane & 15;
  __bf16* pw = Ps + wv * 16 * 64;

  for (int kv0 = 0; kv0 < kv_limit; kv0 += 64) {
    // stage K tile (async) and V tile (transposed scatter)
#pragma unroll
    for (int i = 0; i < 8; ++i) {
      int u = tid + i * 128;
      int r = u >> 4, c = (u & 15) * 8;
      async_load_b128(Ks + r * 128 + c, K + (size_t)(kv0 + r) * D_MODEL + hoff + c);
      uint4 pv = *(const uint4*)(V + (size_t)(kv0 + r) * D_MODEL + hoff + c);
      const __bf16* pe = (const __bf16*)&pv;
#pragma unroll
      for (int j = 0; j < 8; ++j) Vts[(c + j) * 64 + r] = pe[j];
    }
    wait_async0();
    __syncthreads();

    // S = Q K^T  (16 x 64 per wave)
    f32x8 s[4];
#pragma unroll
    for (int nt = 0; nt < 4; ++nt) {
      f32x8 z = {};
      s[nt] = z;
#pragma unroll
      for (int ks = 0; ks < 4; ++ks) {
        bf16x16 bk = load_frag_b(Ks, 128, nt * 16, ks * 32, lane);
        s[nt] = wmma_bf16(aq[ks], bk, s[nt]);
      }
#pragma unroll
      for (int v = 0; v < 8; ++v) s[nt][v] *= scale;
    }

    // online softmax (row stats across 16 lanes of each half + 4 col tiles)
    float cm[8];
#pragma unroll
    for (int v = 0; v < 8; ++v)
      cm[v] = fmaxf(fmaxf(s[0][v], s[1][v]), fmaxf(s[2][v], s[3][v]));
#pragma unroll
    for (int m = 1; m < 16; m <<= 1)
#pragma unroll
      for (int v = 0; v < 8; ++v) cm[v] = fmaxf(cm[v], __shfl_xor(cm[v], m, 32));

    float alpha[8];
#pragma unroll
    for (int v = 0; v < 8; ++v) {
      float mn = fmaxf(m8[v], cm[v]);
      alpha[v] = __expf(m8[v] - mn);
      m8[v] = mn;
    }
#pragma unroll
    for (int nt = 0; nt < 4; ++nt)
#pragma unroll
      for (int v = 0; v < 8; ++v) s[nt][v] = __expf(s[nt][v] - m8[v]);

    float rs[8];
#pragma unroll
    for (int v = 0; v < 8; ++v) rs[v] = s[0][v] + s[1][v] + s[2][v] + s[3][v];
#pragma unroll
    for (int m = 1; m < 16; m <<= 1)
#pragma unroll
      for (int v = 0; v < 8; ++v) rs[v] += __shfl_xor(rs[v], m, 32);
#pragma unroll
    for (int v = 0; v < 8; ++v) l8[v] = l8[v] * alpha[v] + rs[v];

#pragma unroll
    for (int ot = 0; ot < 8; ++ot)
#pragma unroll
      for (int v = 0; v < 8; ++v) acc[ot][v] *= alpha[v];

    // stage P (C-layout -> row-major bf16), wave-private region, DS in-order
#pragma unroll
    for (int nt = 0; nt < 4; ++nt)
#pragma unroll
      for (int v = 0; v < 8; ++v)
        pw[(v + 8 * hg) * 64 + nt * 16 + l15] = (__bf16)s[nt][v];

    // O += P (16x64) x V (64x128)
    bf16x16 pa0 = load_frag_a(pw, 64, 0, 0,  lane);
    bf16x16 pa1 = load_frag_a(pw, 64, 0, 32, lane);
#pragma unroll
    for (int ot = 0; ot < 8; ++ot) {
      bf16x16 bv0 = load_frag_b(Vts, 64, ot * 16, 0,  lane);
      acc[ot] = wmma_bf16(pa0, bv0, acc[ot]);
      bf16x16 bv1 = load_frag_b(Vts, 64, ot * 16, 32, lane);
      acc[ot] = wmma_bf16(pa1, bv1, acc[ot]);
    }
    __syncthreads();
  }

  // normalize and store bf16
  float inv[8];
#pragma unroll
  for (int v = 0; v < 8; ++v) inv[v] = 1.f / l8[v];
#pragma unroll
  for (int ot = 0; ot < 8; ++ot)
#pragma unroll
    for (int v = 0; v < 8; ++v) {
      int row = q0 + wv * 16 + v + 8 * hg;
      Y[(size_t)row * D_MODEL + hoff + ot * 16 + l15] = (__bf16)(acc[ot][v] * inv[v]);
    }
}

// ---------------------------------------------------------------------------
// Elementwise epilogues
// ---------------------------------------------------------------------------
__global__ __launch_bounds__(256) void residual_gated_kernel(
    const float* __restrict__ xin, const float* __restrict__ y,
    const float* __restrict__ modulation, const float* __restrict__ t_mod,
    float* __restrict__ xout, int i_gate) {
  size_t idx = (size_t)blockIdx.x * 256 + threadIdx.x;
  if (idx >= (size_t)T_SEQ * D_MODEL) return;
  int d = (int)(idx % D_MODEL);
  int t = (int)(idx / D_MODEL);
  int f = t >> 9;
  float g = modulation[(size_t)i_gate * D_MODEL + d] +
            t_mod[((size_t)f * 6 + i_gate) * D_MODEL + d];
  xout[idx] = xin[idx] + y[idx] * g;
}

__global__ __launch_bounds__(256) void residual_plain_kernel(
    const float* __restrict__ xin, const float* __restrict__ y,
    float* __restrict__ xout) {
  size_t idx = (size_t)blockIdx.x * 256 + threadIdx.x;
  if (idx >= (size_t)T_SEQ * D_MODEL) return;
  xout[idx] = xin[idx] + y[idx];
}

// ---------------------------------------------------------------------------
// Orchestration
// ---------------------------------------------------------------------------
extern "C" void kernel_launch(void* const* d_in, const int* in_sizes, int n_in,
                              void* d_out, int out_size, void* d_ws, size_t ws_size,
                              hipStream_t stream) {
  (void)in_sizes; (void)n_in; (void)out_size; (void)ws_size;

  const float* x        = (const float*)d_in[0];
  const float* context  = (const float*)d_in[1];
  const float* t_mod    = (const float*)d_in[2];
  const float* fcos     = (const float*)d_in[3];
  const float* fsin     = (const float*)d_in[4];
  const float* sa_wq = (const float*)d_in[5];  const float* sa_bq = (const float*)d_in[6];
  const float* sa_wk = (const float*)d_in[7];  const float* sa_bk = (const float*)d_in[8];
  const float* sa_wv = (const float*)d_in[9];  const float* sa_bv = (const float*)d_in[10];
  const float* sa_wo = (const float*)d_in[11]; const float* sa_bo = (const float*)d_in[12];
  const float* sa_nq = (const float*)d_in[13]; const float* sa_nk = (const float*)d_in[14];
  const float* ca_wq = (const float*)d_in[15]; const float* ca_bq = (const float*)d_in[16];
  const float* ca_wk = (const float*)d_in[17]; const float* ca_bk = (const float*)d_in[18];
  const float* ca_wv = (const float*)d_in[19]; const float* ca_bv = (const float*)d_in[20];
  const float* ca_wo = (const float*)d_in[21]; const float* ca_bo = (const float*)d_in[22];
  const float* ca_nq = (const float*)d_in[23]; const float* ca_nk = (const float*)d_in[24];
  const float* n3_w  = (const float*)d_in[25]; const float* n3_b  = (const float*)d_in[26];
  const float* ffn_w1 = (const float*)d_in[27]; const float* ffn_b1 = (const float*)d_in[28];
  const float* ffn_w2 = (const float*)d_in[29]; const float* ffn_b2 = (const float*)d_in[30];
  const float* modulation = (const float*)d_in[31];
  // d_in[32] = num_frames (hardcoded F=6)

  // --- workspace bump allocator ---
  char* ws = (char*)d_ws;
  size_t off = 0;
  auto alloc = [&](size_t bytes) -> void* {
    void* p = ws + off;
    off = (off + bytes + 255) & ~(size_t)255;
    return p;
  };
  const size_t TD  = (size_t)T_SEQ * D_MODEL;
  const size_t CD  = (size_t)T_CTX * D_MODEL;
  const size_t DD  = (size_t)D_MODEL * D_MODEL;
  const size_t DF  = (size_t)D_MODEL * FFN_DIM;
  const size_t TF  = (size_t)T_SEQ * FFN_DIM;

  __bf16* wt_sa_q = (__bf16*)alloc(DD * 2);
  __bf16* wt_sa_k = (__bf16*)alloc(DD * 2);
  __bf16* wt_sa_v = (__bf16*)alloc(DD * 2);
  __bf16* wt_sa_o = (__bf16*)alloc(DD * 2);
  __bf16* wt_ca_q = (__bf16*)alloc(DD * 2);
  __bf16* wt_ca_k = (__bf16*)alloc(DD * 2);
  __bf16* wt_ca_v = (__bf16*)alloc(DD * 2);
  __bf16* wt_ca_o = (__bf16*)alloc(DD * 2);
  __bf16* wt_f1   = (__bf16*)alloc(DF * 2);   // 8960 x 1536
  __bf16* wt_f2   = (__bf16*)alloc(DF * 2);   // 1536 x 8960
  __bf16* inp_bf  = (__bf16*)alloc(TD * 2);
  float*  qf      = (float*) alloc(TD * 4);
  float*  kf      = (float*) alloc(TD * 4);
  float*  vf      = (float*) alloc(TD * 4);
  __bf16* Qb      = (__bf16*)alloc(TD * 2);
  __bf16* Kb      = (__bf16*)alloc(TD * 2);
  __bf16* Vb      = (__bf16*)alloc(TD * 2);
  __bf16* ybf     = (__bf16*)alloc(TD * 2);
  float*  projf   = (float*) alloc(TD * 4);
  float*  xcur    = (float*) alloc(TD * 4);
  __bf16* ctxbf   = (__bf16*)alloc(CD * 2);
  __bf16* hbf     = (__bf16*)alloc(TF * 2);   // GELU fused into ffn1 epilogue

  auto blocks_for = [](size_t n) { return (unsigned)((n + 255) / 256); };

  // --- pack weights to transposed bf16 (N x K) ---
  transpose_to_bf16<<<blocks_for(DD), 256, 0, stream>>>(sa_wq, wt_sa_q, 1536, 1536);
  transpose_to_bf16<<<blocks_for(DD), 256, 0, stream>>>(sa_wk, wt_sa_k, 1536, 1536);
  transpose_to_bf16<<<blocks_for(DD), 256, 0, stream>>>(sa_wv, wt_sa_v, 1536, 1536);
  transpose_to_bf16<<<blocks_for(DD), 256, 0, stream>>>(sa_wo, wt_sa_o, 1536, 1536);
  transpose_to_bf16<<<blocks_for(DD), 256, 0, stream>>>(ca_wq, wt_ca_q, 1536, 1536);
  transpose_to_bf16<<<blocks_for(DD), 256, 0, stream>>>(ca_wk, wt_ca_k, 1536, 1536);
  transpose_to_bf16<<<blocks_for(DD), 256, 0, stream>>>(ca_wv, wt_ca_v, 1536, 1536);
  transpose_to_bf16<<<blocks_for(DD), 256, 0, stream>>>(ca_wo, wt_ca_o, 1536, 1536);
  transpose_to_bf16<<<blocks_for(DF), 256, 0, stream>>>(ffn_w1, wt_f1, 1536, 8960);
  transpose_to_bf16<<<blocks_for(DF), 256, 0, stream>>>(ffn_w2, wt_f2, 8960, 1536);

  // --- self-attention branch ---
  ln_mod_kernel<<<T_SEQ, 256, 0, stream>>>(x, modulation, t_mod, inp_bf, 0, 1);

  dim3 gproj(T_SEQ / 128, D_MODEL / 128);
  gemm_bf16<0><<<gproj, 256, 0, stream>>>(inp_bf, wt_sa_q, sa_bq, qf, T_SEQ, D_MODEL, D_MODEL);
  gemm_bf16<0><<<gproj, 256, 0, stream>>>(inp_bf, wt_sa_k, sa_bk, kf, T_SEQ, D_MODEL, D_MODEL);
  gemm_bf16<0><<<gproj, 256, 0, stream>>>(inp_bf, wt_sa_v, sa_bv, vf, T_SEQ, D_MODEL, D_MODEL);

  rms_rope_kernel<<<T_SEQ, 256, 0, stream>>>(qf, sa_nq, fcos, fsin, Qb, 1);
  rms_rope_kernel<<<T_SEQ, 256, 0, stream>>>(kf, sa_nk, fcos, fsin, Kb, 1);
  to_bf16_kernel<<<blocks_for(TD), 256, 0, stream>>>(vf, Vb, TD);

  dim3 gattn(T_SEQ / 64, N_HEADS);
  attn_kernel<<<gattn, 128, 0, stream>>>(Qb, Kb, Vb, ybf, T_SEQ, FRAME);

  gemm_bf16<0><<<gproj, 256, 0, stream>>>(ybf, wt_sa_o, sa_bo, projf, T_SEQ, D_MODEL, D_MODEL);
  residual_gated_kernel<<<blocks_for(TD), 256, 0, stream>>>(x, projf, modulation, t_mod, xcur, 2);

  // --- cross-attention branch ---
  ln_affine_kernel<<<T_SEQ, 256, 0, stream>>>(xcur, n3_w, n3_b, inp_bf);
  gemm_bf16<0><<<gproj, 256, 0, stream>>>(inp_bf, wt_ca_q, ca_bq, qf, T_SEQ, D_MODEL, D_MODEL);
  rms_rope_kernel<<<T_SEQ, 256, 0, stream>>>(qf, ca_nq, nullptr, nullptr, Qb, 0);

  to_bf16_kernel<<<blocks_for(CD), 256, 0, stream>>>(context, ctxbf, CD);
  dim3 gctx(T_CTX / 128, D_MODEL / 128);
  gemm_bf16<0><<<gctx, 256, 0, stream>>>(ctxbf, wt_ca_k, ca_bk, kf, T_CTX, D_MODEL, D_MODEL);
  rms_rope_kernel<<<T_CTX, 256, 0, stream>>>(kf, ca_nk, nullptr, nullptr, Kb, 0);
  gemm_bf16<0><<<gctx, 256, 0, stream>>>(ctxbf, wt_ca_v, ca_bv, vf, T_CTX, D_MODEL, D_MODEL);
  to_bf16_kernel<<<blocks_for(CD), 256, 0, stream>>>(vf, Vb, CD);

  attn_kernel<<<gattn, 128, 0, stream>>>(Qb, Kb, Vb, ybf, T_CTX, 0);

  gemm_bf16<0><<<gproj, 256, 0, stream>>>(ybf, wt_ca_o, ca_bo, projf, T_SEQ, D_MODEL, D_MODEL);
  residual_plain_kernel<<<blocks_for(TD), 256, 0, stream>>>(xcur, projf, xcur);

  // --- FFN branch (GELU fused into first GEMM epilogue, bf16 intermediate) ---
  ln_mod_kernel<<<T_SEQ, 256, 0, stream>>>(xcur, modulation, t_mod, inp_bf, 3, 4);
  dim3 gffn1(T_SEQ / 128, FFN_DIM / 128);
  gemm_bf16<1><<<gffn1, 256, 0, stream>>>(inp_bf, wt_f1, ffn_b1, hbf, T_SEQ, FFN_DIM, D_MODEL);
  gemm_bf16<0><<<gproj, 256, 0, stream>>>(hbf, wt_f2, ffn_b2, projf, T_SEQ, D_MODEL, FFN_DIM);
  residual_gated_kernel<<<blocks_for(TD), 256, 0, stream>>>(xcur, projf, modulation, t_mod,
                                                            (float*)d_out, 5);
}
